// TensorProductConv_7275674599677
// MI455X (gfx1250) — matrix operand order repaired and checked by reading, not compiled, not run on hardware
//
#include <hip/hip_runtime.h>

#define NODE_DIM 128
#define NUM_BASIS 8
#define WAVES_E 4
#define ETILES 4          // 16-edge tiles per wave
#define WAVES_N 8
#define NTILES 4          // 16-node tiles per wave

typedef __bf16 v16bf __attribute__((ext_vector_type(16)));
typedef __bf16 v4bf  __attribute__((ext_vector_type(4)));
typedef float  v8f   __attribute__((ext_vector_type(8)));
typedef unsigned int u32x4 __attribute__((ext_vector_type(4)));
typedef unsigned int u32x2 __attribute__((ext_vector_type(2)));

struct Q2 { u32x4 a, b; };

static __device__ __forceinline__ v8f wmma_bf16(v16bf a, v16bf b, v8f c) {
    // v_wmma_f32_16x16x32_bf16: D = A(16x32) * B(32x16) + C
    return __builtin_amdgcn_wmma_f32_16x16x32_bf16(false, a, false, b, (short)0, c, false, false);
}

static __device__ __forceinline__ v8f v8f_zero() {
    v8f c;
#pragma unroll
    for (int i = 0; i < 8; ++i) c[i] = 0.0f;
    return c;
}

static __device__ __forceinline__ v16bf v16bf_zero() {
    v16bf a;
#pragma unroll
    for (int i = 0; i < 16; ++i) a[i] = (__bf16)0.0f;
    return a;
}

// 32 contiguous bytes of LDS -> one v16bf operand (2x ds_load_b128)
static __device__ __forceinline__ v16bf lds_frag(const __bf16* p) {
    Q2 q;
    q.a = *(const u32x4*)p;
    q.b = *(const u32x4*)(p + 8);
    return __builtin_bit_cast(v16bf, q);
}

// two disjoint 16-byte LDS chunks -> one v16bf operand
static __device__ __forceinline__ v16bf lds_frag2(const __bf16* plo, const __bf16* phi) {
    Q2 q;
    q.a = *(const u32x4*)plo;
    q.b = *(const u32x4*)phi;
    return __builtin_bit_cast(v16bf, q);
}

// 16-byte LDS chunk + zero upper half -> v16bf (K-padded fragment)
static __device__ __forceinline__ v16bf lds_frag_lo(const __bf16* plo) {
    Q2 q;
    q.a = *(const u32x4*)plo;
#pragma unroll
    for (int i = 0; i < 4; ++i) q.b[i] = 0u;
    return __builtin_bit_cast(v16bf, q);
}

static __device__ __forceinline__ float silu_fast(float x) {
    return x * __builtin_amdgcn_rcpf(1.0f + __expf(-x));
}

// ---------------------------------------------------------------------------
// Kernel 0: zero the aggregation workspace
// ---------------------------------------------------------------------------
__global__ void zero_kernel(float* __restrict__ p, long long n) {
    long long i = (long long)blockIdx.x * blockDim.x + threadIdx.x;
    long long stride = (long long)gridDim.x * blockDim.x;
    for (; i < n; i += stride) p[i] = 0.0f;
}

// ---------------------------------------------------------------------------
// Kernel 1: fold the output projection:
//   Wfused = Wself @ Wcat[0:128], bfused = bself @ Wcat[0:128] + bcat
// ---------------------------------------------------------------------------
__global__ __launch_bounds__(256) void fuse_kernel(
    const float* __restrict__ Wself, const float* __restrict__ bself,
    const float* __restrict__ Wcat,  const float* __restrict__ bcat,
    float* __restrict__ wfused, float* __restrict__ bfused)
{
    const int tid = threadIdx.x;
    for (int i = tid; i < NODE_DIM * NODE_DIM; i += 256) {
        const int r = i >> 7, c = i & 127;
        float s = 0.0f;
        for (int t = 0; t < NODE_DIM; ++t)
            s += Wself[r * NODE_DIM + t] * Wcat[t * NODE_DIM + c];
        wfused[i] = s;
    }
    for (int c = tid; c < NODE_DIM; c += 256) {
        float s = bcat[c];
        for (int t = 0; t < NODE_DIM; ++t)
            s += bself[t] * Wcat[t * NODE_DIM + c];
        bfused[c] = s;
    }
}

// ---------------------------------------------------------------------------
// Kernel 2: per-edge radial MLP (WMMA), row-coalesced gather, atomic scatter
// One wave processes ETILES consecutive 16-edge tiles; stage weights once.
// ---------------------------------------------------------------------------
__global__ __launch_bounds__(32 * WAVES_E) void edge_kernel(
    const float* __restrict__ nf, const int* __restrict__ ei,
    const float* __restrict__ rbf,
    const float* __restrict__ W1, const float* __restrict__ b1,
    const float* __restrict__ W2, const float* __restrict__ b2,
    float* __restrict__ agg, int E, int N)
{
    __shared__ __align__(16) __bf16 w1B[8][32][8];              // 4 KB, B-frag order
    __shared__ float  b1s[NODE_DIM], b2s[NODE_DIM];             // 1 KB
    __shared__ __align__(16) __bf16 w2B[4][8][32][16];          // 32 KB, B-frag order
    // hA: first holds layer-1 activations h, then reused for the edge weights
    __shared__ __align__(16) __bf16 hA[WAVES_E][16][NODE_DIM];  // 16 KB
    __shared__ int    eRow[WAVES_E][16], eCol[WAVES_E][16];

    const int tid  = threadIdx.x;
    const int wv   = tid >> 5;
    const int lane = tid & 31;
    const int g    = lane >> 4;
    const int n16  = lane & 15;
    const int m    = n16;            // A-matrix row owned by this lane

    // ---- stage weights/biases into LDS (coalesced global reads) ----
    for (int i = tid; i < NODE_DIM; i += blockDim.x) { b1s[i] = b1[i]; b2s[i] = b2[i]; }
    // W1 (8x128) -> B-frag layout, lanes >=16 hold zeros (K padded 8 -> 32)
    for (int i = tid; i < 8 * 32 * 8; i += blockDim.x) {
        const int e = i & 7, l = (i >> 3) & 31, nt = i >> 8;
        w1B[nt][l][e] = (l < 16) ? (__bf16)W1[e * NODE_DIM + nt * 16 + l] : (__bf16)0.0f;
    }
    // W2 (128x128) -> B-frag layout; iterate row-major with float4 loads
    for (int i4 = tid * 4; i4 < NODE_DIM * NODE_DIM; i4 += blockDim.x * 4) {
        const int k = i4 >> 7, n0 = i4 & 127;
        const float4 w = *(const float4*)&W2[i4];
        const int kt = k >> 5, e = (k & 7) | ((k >> 1) & 8), lg = ((k >> 3) & 1) * 16;
        const float wv4[4] = {w.x, w.y, w.z, w.w};
#pragma unroll
        for (int d = 0; d < 4; ++d) {
            const int n = n0 + d;
            w2B[kt][n >> 4][lg + (n & 15)][e] = (__bf16)wv4[d];
        }
    }
    __syncthreads();   // only barrier: cross-wave weight staging

    const long long waveBase =
        ((long long)blockIdx.x * WAVES_E + wv) * (16 * ETILES);

    for (int tt = 0; tt < ETILES; ++tt) {
        const long long tileBase = waveBase + tt * 16;

        if (lane < 16) {                     // wave-private, LDS in-order per wave
            const long long e = tileBase + lane;
            eRow[wv][lane] = (e < E) ? ei[e] : 0;
            eCol[wv][lane] = (e < E) ? ei[(long long)E + e] : 0;
        }

        // ---- layer 1: h = silu(rbf @ W1 + b1), K = 8 zero-padded to 32 ----
        v16bf aR = v16bf_zero();
        {
            const long long eMy = tileBase + m;
            if (g == 0 && eMy < E) {
                const float4* rp = (const float4*)(rbf + eMy * NUM_BASIS);
                const float4 r0 = rp[0], r1 = rp[1];
                aR[0] = (__bf16)r0.x; aR[1] = (__bf16)r0.y;
                aR[2] = (__bf16)r0.z; aR[3] = (__bf16)r0.w;
                aR[4] = (__bf16)r1.x; aR[5] = (__bf16)r1.y;
                aR[6] = (__bf16)r1.z; aR[7] = (__bf16)r1.w;
            }
        }
#pragma unroll
        for (int nt = 0; nt < 8; ++nt) {
            v8f c = wmma_bf16(aR, lds_frag_lo(&w1B[nt][lane][0]), v8f_zero());
            const int colc = nt * 16 + n16;
#pragma unroll
            for (int r = 0; r < 8; ++r) {
                const int row = r + g * 8;
                hA[wv][row][colc] = (__bf16)silu_fast(c[r] + b1s[colc]);
            }
        }

        // ---- layer 2: weights = h @ W2 + b2 ----
        v16bf aH[4];
#pragma unroll
        for (int kt = 0; kt < 4; ++kt)
            aH[kt] = lds_frag2(&hA[wv][m][kt * 32 + g * 8],
                               &hA[wv][m][kt * 32 + 16 + g * 8]);
#pragma unroll
        for (int nt = 0; nt < 8; ++nt) {
            v8f acc = v8f_zero();
#pragma unroll
            for (int kt = 0; kt < 4; ++kt)
                acc = wmma_bf16(aH[kt], lds_frag(&w2B[kt][nt][lane][0]), acc);
            const int colc = nt * 16 + n16;
#pragma unroll
            for (int r = 0; r < 8; ++r) {
                const int row = r + g * 8;
                hA[wv][row][colc] = (__bf16)(acc[r] + b2s[colc]);  // reuse as weights
            }
        }

        // ---- gather / multiply / scatter: one edge row at a time, coalesced ----
        for (int t = 0; t < 16; ++t) {
            const long long e = tileBase + t;   // wave-uniform guard
            if (e < E) {
                const int j   = eCol[wv][t];
                const int dst = eRow[wv][t];
                const float4 x = *(const float4*)&nf[(long long)j * NODE_DIM + lane * 4];
                const v4bf w4 = __builtin_bit_cast(v4bf, *(const u32x2*)&hA[wv][t][lane * 4]);
                float* ap = &agg[(long long)dst * NODE_DIM + lane * 4];
                atomicAdd(ap + 0, (float)w4[0] * x.x);
                atomicAdd(ap + 1, (float)w4[1] * x.y);
                atomicAdd(ap + 2, (float)w4[2] * x.z);
                atomicAdd(ap + 3, (float)w4[3] * x.w);
            }
        }
    }
}

// ---------------------------------------------------------------------------
// Kernel 3: out = [nf | agg] @ [Wfused ; WcatBot] + bfused
// One wave processes NTILES 16-node tiles, K = 256 (8 K-tiles).
// ---------------------------------------------------------------------------
__global__ __launch_bounds__(32 * WAVES_N) void node_kernel(
    const float* __restrict__ nf, const float* __restrict__ agg,
    const float* __restrict__ Wcat, const float* __restrict__ wfused,
    const float* __restrict__ bfused, float* __restrict__ out, int N)
{
    __shared__ __align__(16) __bf16 bComb[8][8][32][16];   // exactly 64 KB

    const int tid  = threadIdx.x;
    const int wv   = tid >> 5;
    const int lane = tid & 31;
    const int g    = lane >> 4;
    const int n16  = lane & 15;
    const int m    = n16;

    // Stage combined 256x128 B (rows 0..127 = Wfused, 128..255 = Wcat bottom),
    // iterating row-major with float4 loads for coalescing.
    for (int i4 = tid * 4; i4 < 2 * NODE_DIM * NODE_DIM; i4 += blockDim.x * 4) {
        const int k = i4 >> 7, n0 = i4 & 127;
        const float4 w = (k < NODE_DIM) ? *(const float4*)&wfused[i4]
                                        : *(const float4*)&Wcat[i4];
        const int kt = k >> 5, e = (k & 7) | ((k >> 1) & 8), lg = ((k >> 3) & 1) * 16;
        const float wv4[4] = {w.x, w.y, w.z, w.w};
#pragma unroll
        for (int d = 0; d < 4; ++d) {
            const int n = n0 + d;
            bComb[kt][n >> 4][lg + (n & 15)][e] = (__bf16)wv4[d];
        }
    }
    __syncthreads();

    const int waveBase = (blockIdx.x * WAVES_N + wv) * (16 * NTILES);

    for (int tt = 0; tt < NTILES; ++tt) {
        const int tb   = waveBase + tt * 16;
        const int node = tb + m;
        const long long safe = (long long)((node < N) ? node : 0) * NODE_DIM;
        const float* arow = nf  + safe;
        const float* grow = agg + safe;

        v16bf aF[8];
#pragma unroll
        for (int kt = 0; kt < 8; ++kt) {
            const float* src = (kt < 4) ? arow : grow;
            const int k0 = (kt & 3) * 32 + g * 8;
            const float4 lo0 = *(const float4*)&src[k0];
            const float4 lo1 = *(const float4*)&src[k0 + 4];
            const float4 hi0 = *(const float4*)&src[k0 + 16];
            const float4 hi1 = *(const float4*)&src[k0 + 20];
            aF[kt][0]  = (__bf16)lo0.x; aF[kt][1]  = (__bf16)lo0.y;
            aF[kt][2]  = (__bf16)lo0.z; aF[kt][3]  = (__bf16)lo0.w;
            aF[kt][4]  = (__bf16)lo1.x; aF[kt][5]  = (__bf16)lo1.y;
            aF[kt][6]  = (__bf16)lo1.z; aF[kt][7]  = (__bf16)lo1.w;
            aF[kt][8]  = (__bf16)hi0.x; aF[kt][9]  = (__bf16)hi0.y;
            aF[kt][10] = (__bf16)hi0.z; aF[kt][11] = (__bf16)hi0.w;
            aF[kt][12] = (__bf16)hi1.x; aF[kt][13] = (__bf16)hi1.y;
            aF[kt][14] = (__bf16)hi1.z; aF[kt][15] = (__bf16)hi1.w;
        }
#pragma unroll
        for (int nt = 0; nt < 8; ++nt) {
            v8f acc = v8f_zero();
#pragma unroll
            for (int kt = 0; kt < 8; ++kt)
                acc = wmma_bf16(aF[kt], lds_frag(&bComb[kt][nt][lane][0]), acc);
            const int colc = nt * 16 + n16;
#pragma unroll
            for (int r = 0; r < 8; ++r) {
                const int row = r + g * 8;
                const int nd  = tb + row;
                if (nd < N) out[(long long)nd * NODE_DIM + colc] = acc[r] + bfused[colc];
            }
        }
    }
}

// ---------------------------------------------------------------------------
extern "C" void kernel_launch(void* const* d_in, const int* in_sizes, int n_in,
                              void* d_out, int out_size, void* d_ws, size_t ws_size,
                              hipStream_t stream)
{
    const float* nf    = (const float*)d_in[0];
    const int*   ei    = (const int*)d_in[1];    // edge_index [2,E]
    const float* rbf   = (const float*)d_in[2];  // [E,8]
    // d_in[3] edge_sh unused (lmax = 0)
    const float* W1    = (const float*)d_in[4];
    const float* b1    = (const float*)d_in[5];
    const float* W2    = (const float*)d_in[6];
    const float* b2    = (const float*)d_in[7];
    const float* Wself = (const float*)d_in[8];
    const float* bself = (const float*)d_in[9];
    const float* Wcat  = (const float*)d_in[10];
    const float* bcat  = (const float*)d_in[11];
    float* out = (float*)d_out;

    const int N = in_sizes[0] / NODE_DIM;
    const int E = in_sizes[2] / NUM_BASIS;

    float* agg    = (float*)d_ws;                          // N*128 floats
    float* wfused = agg + (long long)N * NODE_DIM;         // 128*128 floats
    float* bfused = wfused + NODE_DIM * NODE_DIM;          // 128 floats

    const long long aggN = (long long)N * NODE_DIM;
    zero_kernel<<<1024, 256, 0, stream>>>(agg, aggN);
    fuse_kernel<<<1, 256, 0, stream>>>(Wself, bself, Wcat, bcat, wfused, bfused);

    const int edgesPerBlock = WAVES_E * 16 * ETILES;   // 256
    const int edgeBlocks = (E + edgesPerBlock - 1) / edgesPerBlock;
    edge_kernel<<<edgeBlocks, 32 * WAVES_E, 0, stream>>>(nf, ei, rbf, W1, b1, W2, b2, agg, E, N);

    const int nodesPerBlock = WAVES_N * 16 * NTILES;   // 512
    const int nodeBlocks = (N + nodesPerBlock - 1) / nodesPerBlock;
    node_kernel<<<nodeBlocks, 32 * WAVES_N, 0, stream>>>(nf, agg, Wcat, wfused, bfused, out, N);
}
